// DGDAGRNN_58763742544948
// MI455X (gfx1250) — compile-verified, matrix operand order checked by reading, not run on hardware
//
#include <hip/hip_runtime.h>
#include <hip/hip_bf16.h>
#include <stdint.h>

// Problem constants (from setup_inputs): L=8, S=8192, F=2, R=3, vhs=100, nvt=3
#define L_  8
#define S_  8192
#define F_  2
#define R_  3
#define N_  (L_*S_)
#define VH  100
#define VHP 128        // padded K (bf16: 64 dwords per row)
#define G3  300
#define G3P 304        // padded gate width (19 n-tiles)
#define NT_G 19
#define NT_M 7         // 112 cols for 100-wide message GEMMs

typedef __attribute__((ext_vector_type(16))) __bf16    v16bf;
typedef __attribute__((ext_vector_type(2)))  __bf16    v2bf;
typedef __attribute__((ext_vector_type(8)))  float     v8f;
typedef __attribute__((ext_vector_type(4)))  uint32_t  v4u;
typedef __attribute__((ext_vector_type(8)))  int32_t   v8i_;
typedef __attribute__((ext_vector_type(4)))  int32_t   v4i_;

union FragU { v16bf v; uint32_t u[8]; };

// f32 pair -> packed bf16 dword using the native converter (v_cvt_pk_bf16_f32),
// instead of integer RNE emulation (which bloated VALU with v_add3/v_and/v_mov_b16).
__device__ __forceinline__ uint32_t pack_bf2(float a, float b) {
#if __has_builtin(__builtin_amdgcn_cvt_pk_bf16_f32)
  v2bf p = __builtin_amdgcn_cvt_pk_bf16_f32(a, b);
  union { v2bf h; uint32_t u; } r; r.h = p;
  return r.u;
#else
  union { __bf16 h[2]; uint32_t u; } r;
  r.h[0] = (__bf16)a; r.h[1] = (__bf16)b;   // lowers to v_cvt_*_bf16_f32
  return r.u;
#endif
}

// A fragment (16x32 bf16) straight from global f32 rows into registers.
// CDNA5 layout: lanes 0-15 hold row(lane), k {kt*32+0..7, kt*32+16..23};
//               lanes 16-31 hold row(lane-16), k {+8..15, +24..31}.
__device__ __forceinline__ v16bf load_a_global(const float* __restrict__ rowp, int kt, int hi) {
  const float* p = rowp + kt * 32 + hi * 8;
  float4 a0 = *(const float4*)(p);
  float4 a1 = *(const float4*)(p + 4);
  float4 b0 = *(const float4*)(p + 16);
  float4 b1 = *(const float4*)(p + 20);
  FragU f;
  f.u[0] = pack_bf2(a0.x, a0.y); f.u[1] = pack_bf2(a0.z, a0.w);
  f.u[2] = pack_bf2(a1.x, a1.y); f.u[3] = pack_bf2(a1.z, a1.w);
  f.u[4] = pack_bf2(b0.x, b0.y); f.u[5] = pack_bf2(b0.z, b0.w);
  f.u[6] = pack_bf2(b1.x, b1.y); f.u[7] = pack_bf2(b1.z, b1.w);
  return f.v;
}

// B fragment (32x16, B = W^T) from a row-major [rows][128 bf16] LDS tile.
__device__ __forceinline__ v16bf load_b_lds(const uint32_t* base, int lane, int kt) {
  int r = lane & 15, hi = lane >> 4;
  const uint32_t* p = base + r * 64 + kt * 16 + hi * 4;
  FragU f;
  #pragma unroll
  for (int i = 0; i < 4; ++i) { f.u[i] = p[i]; f.u[4 + i] = p[8 + i]; }
  return f.v;
}

// Stage a prepacked-bf16 weight chunk (ndw dwords, 64 dwords per output-neuron row)
// into LDS: TDM descriptor-driven async prefetch from wave 0 (tracked by TENSORcnt),
// then an authoritative cooperative b128 copy. 256 threads per block.
__device__ __forceinline__ void stage_w_lds(uint32_t* ldsw, const uint32_t* __restrict__ W, int ndw) {
  __syncthreads();                      // previous chunk consumers done
#if __has_builtin(__builtin_amdgcn_tensor_load_to_lds)
  if (threadIdx.x < 32) {
    int rows = ndw >> 6;
    uint64_t ga = (uint64_t)(uintptr_t)W;
    v4u  g0; v8i_ g1; v4i_ g2, g3; v8i_ g4;
    g0[0] = 1u;                                            // count=1 (valid user D#)
    g0[1] = (uint32_t)(uintptr_t)ldsw;                     // lds_addr
    g0[2] = (uint32_t)ga;                                  // global_addr[95:64]
    g0[3] = (uint32_t)((ga >> 32) & 0x1FFFFFFu) | (2u << 30); // addr msbs | type=2
    g1[0] = 0x10000;                                       // data_size=2B, no multicast
    g1[1] = (128u & 0xFFFFu) << 16;                        // tensor_dim0 = 128 elems
    g1[2] = ((uint32_t)rows & 0xFFFFu) << 16;              // tensor_dim1 = rows
    g1[3] = (128u & 0xFFFFu) << 16;                        // tile_dim0 = 128
    g1[4] = (uint32_t)rows & 0xFFFFu;                      // tile_dim1 = rows
    g1[5] = 128u;                                          // tensor_dim0_stride = 128
    g1[6] = (128u & 0xFFFFu) << 16;                        // tensor_dim1_stride = 128
    g1[7] = 0;
    g2[0] = g2[1] = g2[2] = g2[3] = 0;
    g3[0] = g3[1] = g3[2] = g3[3] = 0;
    g4[0] = g4[1] = g4[2] = g4[3] = g4[4] = g4[5] = g4[6] = g4[7] = 0;
    // clang-23 / therock-10.0 arity: (v4u, v8i, v4i, v4i, v8i, i32 cpol)
    __builtin_amdgcn_tensor_load_to_lds(g0, g1, g2, g3, g4, 0);
    __builtin_amdgcn_s_wait_tensorcnt(0);                  // s_wait_tensorcnt 0
  }
#endif
  __syncthreads();
  for (int i = threadIdx.x; i < (ndw >> 2); i += 256)      // authoritative fill
    ((uint4*)ldsw)[i] = ((const uint4*)W)[i];
  __syncthreads();
}

// ---------------- C = A(MxK=128) * W^T; W prepacked bf16 [NTILES*16 rows][128] ------------------
template<int NTILES, int LDC>
__global__ __launch_bounds__(256) void gemm_bf16(const float* __restrict__ A, int M,
                                                 const uint32_t* __restrict__ W,
                                                 float* __restrict__ C) {
  __shared__ uint32_t ldsW[8 * 1024];                      // 8-tile chunk (32 KB)
  int lane = threadIdx.x & 31, wave = threadIdx.x >> 5;
  int m0 = blockIdx.x * 128 + wave * 16;                   // grids are exact: m0+16 <= M
  int r15 = lane & 15, hi = lane >> 4;
  const float* rowp = A + (long)(m0 + r15) * VHP;
  v16bf a0 = load_a_global(rowp, 0, hi);
  v16bf a1 = load_a_global(rowp, 1, hi);
  v16bf a2 = load_a_global(rowp, 2, hi);
  v16bf a3 = load_a_global(rowp, 3, hi);
  float* crow = C + (long)(m0 + hi * 8) * LDC;
  for (int c0 = 0; c0 < NTILES; c0 += 8) {
    int nct = (NTILES - c0 < 8) ? (NTILES - c0) : 8;
    stage_w_lds(ldsW, W + c0 * 1024, nct * 1024);
    for (int nt = 0; nt < nct; ++nt) {
      const uint32_t* wt = ldsW + nt * 1024;
      v8f acc = {};
      acc = __builtin_amdgcn_wmma_f32_16x16x32_bf16(false, a0, false, load_b_lds(wt, lane, 0), (short)0, acc, false, false);
      acc = __builtin_amdgcn_wmma_f32_16x16x32_bf16(false, a1, false, load_b_lds(wt, lane, 1), (short)0, acc, false, false);
      acc = __builtin_amdgcn_wmma_f32_16x16x32_bf16(false, a2, false, load_b_lds(wt, lane, 2), (short)0, acc, false, false);
      acc = __builtin_amdgcn_wmma_f32_16x16x32_bf16(false, a3, false, load_b_lds(wt, lane, 3), (short)0, acc, false, false);
      int n = (c0 + nt) * 16 + r15;
      #pragma unroll
      for (int rr = 0; rr < 8; ++rr)                       // LDC compile-time -> offset stores
        crow[rr * LDC + n] = acc[rr];
    }
  }
}

// -------- fused message kernel: gather rows, dual GEMM (gate+msg), sigmoid*mul, atomic scatter ---
__global__ __launch_bounds__(256) void msg_bf16(const float* __restrict__ h,
                                                const int* __restrict__ gsrc,
                                                const int* __restrict__ sdst, int E,
                                                const uint32_t* __restrict__ Wg,
                                                const uint32_t* __restrict__ Wm,
                                                const float* __restrict__ bg,
                                                float* __restrict__ ps, int lo) {
  __shared__ uint32_t ldsG[NT_M * 1024];                   // 28 KB
  __shared__ uint32_t ldsM[NT_M * 1024];                   // 28 KB
  int lane = threadIdx.x & 31, wave = threadIdx.x >> 5;
  int m0 = blockIdx.x * 128 + wave * 16;
  int r15 = lane & 15, hi = lane >> 4;
  const float* rowp = h + (long)gsrc[m0 + r15] * VHP;      // gathered activation row
  v16bf a0 = load_a_global(rowp, 0, hi);
  v16bf a1 = load_a_global(rowp, 1, hi);
  v16bf a2 = load_a_global(rowp, 2, hi);
  v16bf a3 = load_a_global(rowp, 3, hi);
  stage_w_lds(ldsG, Wg, NT_M * 1024);
  stage_w_lds(ldsM, Wm, NT_M * 1024);
  for (int nt = 0; nt < NT_M; ++nt) {
    const uint32_t* wg = ldsG + nt * 1024;
    const uint32_t* wm = ldsM + nt * 1024;
    v8f g = {}, m = {};
    g = __builtin_amdgcn_wmma_f32_16x16x32_bf16(false, a0, false, load_b_lds(wg, lane, 0), (short)0, g, false, false);
    m = __builtin_amdgcn_wmma_f32_16x16x32_bf16(false, a0, false, load_b_lds(wm, lane, 0), (short)0, m, false, false);
    g = __builtin_amdgcn_wmma_f32_16x16x32_bf16(false, a1, false, load_b_lds(wg, lane, 1), (short)0, g, false, false);
    m = __builtin_amdgcn_wmma_f32_16x16x32_bf16(false, a1, false, load_b_lds(wm, lane, 1), (short)0, m, false, false);
    g = __builtin_amdgcn_wmma_f32_16x16x32_bf16(false, a2, false, load_b_lds(wg, lane, 2), (short)0, g, false, false);
    m = __builtin_amdgcn_wmma_f32_16x16x32_bf16(false, a2, false, load_b_lds(wm, lane, 2), (short)0, m, false, false);
    g = __builtin_amdgcn_wmma_f32_16x16x32_bf16(false, a3, false, load_b_lds(wg, lane, 3), (short)0, g, false, false);
    m = __builtin_amdgcn_wmma_f32_16x16x32_bf16(false, a3, false, load_b_lds(wm, lane, 3), (short)0, m, false, false);
    int n = nt * 16 + r15;
    if (n < VH) {
      float bgn = bg[n];
      #pragma unroll
      for (int rr = 0; rr < 8; ++rr) {
        int e = m0 + hi * 8 + rr;
        float val = m[rr] / (1.f + __expf(-(g[rr] + bgn)));   // sigmoid(g+bg)*m
        int d = sdst[e] - lo;
        atomicAdd(ps + (long)d * VHP + n, val);               // global_atomic_add_f32
      }
    }
  }
}

// ---------------- GRU epilogues (elementwise; fwd gi has K=3 so done scalar) ---------------------
__global__ __launch_bounds__(256) void gru_fwd(const float* __restrict__ inp,   // S x 3
                                               const float* __restrict__ Wi,    // 300 x 3
                                               const float* __restrict__ bi,
                                               const float* __restrict__ bh,
                                               const float* __restrict__ ghb,   // S x 304
                                               const float* __restrict__ psb,   // S x 128
                                               float* __restrict__ hrow) {      // S x 128 (+=)
  int tid = blockIdx.x * 256 + threadIdx.x;
  int row = tid >> 7, n = tid & 127;
  if (n >= VH) return;
  float x0 = inp[row * 3], x1 = inp[row * 3 + 1], x2 = inp[row * 3 + 2];
  float gir = bi[n]          + x0 * Wi[n * 3]            + x1 * Wi[n * 3 + 1]            + x2 * Wi[n * 3 + 2];
  float giz = bi[VH + n]     + x0 * Wi[(VH + n) * 3]     + x1 * Wi[(VH + n) * 3 + 1]     + x2 * Wi[(VH + n) * 3 + 2];
  float gin = bi[2 * VH + n] + x0 * Wi[(2 * VH + n) * 3] + x1 * Wi[(2 * VH + n) * 3 + 1] + x2 * Wi[(2 * VH + n) * 3 + 2];
  long ro = (long)row * G3P;
  float ghr = ghb[ro + n] + bh[n];
  float ghz = ghb[ro + VH + n] + bh[VH + n];
  float ghn = ghb[ro + 2 * VH + n] + bh[2 * VH + n];
  float r = 1.f / (1.f + __expf(-(gir + ghr)));
  float z = 1.f / (1.f + __expf(-(giz + ghz)));
  float nn = tanhf(gin + r * ghn);
  float hp = psb[(long)row * VHP + n];
  hrow[(long)row * VHP + n] += (1.f - z) * nn + z * hp;
}

__global__ __launch_bounds__(256) void gru_bwd(const float* __restrict__ gib,   // S x 304
                                               const float* __restrict__ bi,
                                               const float* __restrict__ bh,
                                               const float* __restrict__ ghb,   // S x 304
                                               const float* __restrict__ psb,
                                               float* __restrict__ hrow) {
  int tid = blockIdx.x * 256 + threadIdx.x;
  int row = tid >> 7, n = tid & 127;
  if (n >= VH) return;
  long ro = (long)row * G3P;
  float gir = gib[ro + n] + bi[n];
  float giz = gib[ro + VH + n] + bi[VH + n];
  float gin = gib[ro + 2 * VH + n] + bi[2 * VH + n];
  float ghr = ghb[ro + n] + bh[n];
  float ghz = ghb[ro + VH + n] + bh[VH + n];
  float ghn = ghb[ro + 2 * VH + n] + bh[2 * VH + n];
  float r = 1.f / (1.f + __expf(-(gir + ghr)));
  float z = 1.f / (1.f + __expf(-(giz + ghz)));
  float nn = tanhf(gin + r * ghn);
  float hp = psb[(long)row * VHP + n];
  hrow[(long)row * VHP + n] += (1.f - z) * nn + z * hp;
}

// ---------------- small utility kernels ----------------------------------------------------------
__global__ __launch_bounds__(256) void pack_w(const float* __restrict__ W, int rows, int cols,
                                              uint32_t* __restrict__ dst, int rp) {
  int tid = blockIdx.x * 256 + threadIdx.x;
  if (tid >= rp * 64) return;
  int row = tid >> 6, j = tid & 63, k0 = j * 2;
  float a = (row < rows && k0     < cols) ? W[(long)row * cols + k0]     : 0.f;
  float b = (row < rows && k0 + 1 < cols) ? W[(long)row * cols + k0 + 1] : 0.f;
  dst[tid] = pack_bf2(a, b);
}

__global__ __launch_bounds__(256) void fillz(float* __restrict__ p, long n) {
  long t = (long)blockIdx.x * 256 + threadIdx.x;
  if (t < n) p[t] = 0.f;
}

// Reference zeroes h BEFORE inp_all = h@Wp.T + bp, so for rounds>0 inp_all == bp broadcast.
__global__ __launch_bounds__(256) void fill_bp(float* __restrict__ p, const float* __restrict__ bp, int n3) {
  int t = blockIdx.x * 256 + threadIdx.x;
  if (t < n3) p[t] = bp[t % 3];
}

__global__ __launch_bounds__(256) void copy_out(float* __restrict__ out, const float* __restrict__ h) {
  int t = blockIdx.x * 256 + threadIdx.x;
  if (t < N_ * VH) {
    int i = t / VH, n = t - i * VH;
    out[t] = h[(long)i * VHP + n];
  }
}

// -------------------------------------------------------------------------------------------------
extern "C" void kernel_launch(void* const* d_in, const int* in_sizes, int n_in,
                              void* d_out, int out_size, void* d_ws, size_t ws_size,
                              hipStream_t stream) {
  const float* x    = (const float*)d_in[0];
  const float* Wi_f = (const float*)d_in[1];
  const float* Wh_f = (const float*)d_in[2];
  const float* bi_f = (const float*)d_in[3];
  const float* bh_f = (const float*)d_in[4];
  const float* Wi_b = (const float*)d_in[5];
  const float* Wh_b = (const float*)d_in[6];
  const float* bi_b = (const float*)d_in[7];
  const float* bh_b = (const float*)d_in[8];
  const float* bp   = (const float*)d_in[10];
  const float* Wg_f = (const float*)d_in[11];
  const float* bg_f = (const float*)d_in[12];
  const float* Wm_f = (const float*)d_in[13];
  const float* Wg_b = (const float*)d_in[14];
  const float* bg_b = (const float*)d_in[15];
  const float* Wm_b = (const float*)d_in[16];
  const int*   esrc = (const int*)d_in[17];
  const int*   edst = (const int*)d_in[18];
  float* out = (float*)d_out;

  // workspace layout
  float* h   = (float*)d_ws;                      // N x 128
  float* ps  = h   + (size_t)N_ * VHP;            // S x 128
  float* ghb = ps  + (size_t)S_ * VHP;            // S x 304
  float* gib = ghb + (size_t)S_ * G3P;            // S x 304
  float* ina = gib + (size_t)S_ * G3P;            // N x 3
  uint32_t* Whf = (uint32_t*)(ina + (size_t)N_ * 3);
  uint32_t* Whb = Whf + G3P * 64;
  uint32_t* Wib = Whb + G3P * 64;
  uint32_t* Wgf = Wib + G3P * 64;
  uint32_t* Wmf = Wgf + 112 * 64;
  uint32_t* Wgb = Wmf + 112 * 64;
  uint32_t* Wmb = Wgb + 112 * 64;

  dim3 B(256);
  int gw3 = (G3P * 64 + 255) / 256, gw1 = (112 * 64 + 255) / 256;
  pack_w<<<gw3, B, 0, stream>>>(Wh_f, G3, VH, Whf, G3P);
  pack_w<<<gw3, B, 0, stream>>>(Wh_b, G3, VH, Whb, G3P);
  pack_w<<<gw3, B, 0, stream>>>(Wi_b, G3, VH, Wib, G3P);
  pack_w<<<gw1, B, 0, stream>>>(Wg_f, VH, VH, Wgf, 112);
  pack_w<<<gw1, B, 0, stream>>>(Wm_f, VH, VH, Wmf, 112);
  pack_w<<<gw1, B, 0, stream>>>(Wg_b, VH, VH, Wgb, 112);
  pack_w<<<gw1, B, 0, stream>>>(Wm_b, VH, VH, Wmb, 112);

  const int E = S_ * F_;
  for (int r = 0; r < R_; ++r) {
    fillz<<<(N_ * VHP) / 256, B, 0, stream>>>(h, (long)N_ * VHP);
    const float* inp_base = x;
    if (r > 0) {
      fill_bp<<<(N_ * 3 + 255) / 256, B, 0, stream>>>(ina, bp, N_ * 3);
      inp_base = ina;
    }
    // forward sweep
    for (int l = 0; l < L_; ++l) {
      int lo = l * S_;
      fillz<<<(S_ * VHP) / 256, B, 0, stream>>>(ps, (long)S_ * VHP);
      if (l > 0) {
        int off = (l - 1) * E;
        msg_bf16<<<E / 128, B, 0, stream>>>(h, esrc + off, edst + off, E, Wgf, Wmf, bg_f, ps, lo);
      }
      gemm_bf16<NT_G, G3P><<<S_ / 128, B, 0, stream>>>(ps, S_, Whf, ghb);
      gru_fwd<<<(S_ * VHP) / 256, B, 0, stream>>>(inp_base + (size_t)lo * 3, Wi_f, bi_f, bh_f,
                                                  ghb, ps, h + (size_t)lo * VHP);
    }
    // backward sweep
    for (int bi2 = 0; bi2 < L_; ++bi2) {
      int f = L_ - 1 - bi2;
      int lo = f * S_;
      fillz<<<(S_ * VHP) / 256, B, 0, stream>>>(ps, (long)S_ * VHP);
      if (bi2 > 0) {
        int off = f * E;
        msg_bf16<<<E / 128, B, 0, stream>>>(h, edst + off, esrc + off, E, Wgb, Wmb, bg_b, ps, lo);
      }
      gemm_bf16<NT_G, G3P><<<S_ / 128, B, 0, stream>>>(h + (size_t)lo * VHP, S_, Wib, gib);
      gemm_bf16<NT_G, G3P><<<S_ / 128, B, 0, stream>>>(ps, S_, Whb, ghb);
      gru_bwd<<<(S_ * VHP) / 256, B, 0, stream>>>(gib, bi_b, bh_b, ghb, ps, h + (size_t)lo * VHP);
    }
    copy_out<<<(N_ * VH + 255) / 256, B, 0, stream>>>(out + (size_t)r * N_ * VH, h);
  }
}
